// LightMultiHeadAttention_62268435858121
// MI455X (gfx1250) — matrix operand order repaired and checked by reading, not compiled, not run on hardware
//
#include <hip/hip_runtime.h>
#include <hip/hip_bf16.h>

// ---------------------------------------------------------------------------
// LightMultiHeadAttention for MI455X (gfx1250), bf16 WMMA pipeline with
// async global->LDS tile copies (ASYNCcnt) and double-buffered LDS.
//   B=4, S=2048, E=1024
//   q = query@Wq^T + bq ; k = key@Wk^T + bk
//   attn = softmax(q k^T / 32)              (attn is an OUTPUT -> materialize)
//   av = attn @ value
//   out = scramble(av) @ Wo^T + bo          (scramble = transpose(0,2,1).view)
// ---------------------------------------------------------------------------

using bf16 = __bf16;
typedef __attribute__((ext_vector_type(16))) __bf16 v16bf;
typedef __attribute__((ext_vector_type(8)))  float  v8f;
typedef int v4i_vs __attribute__((vector_size(16)));   // int4, gcc-vector style

static constexpr int BDIM = 4;
static constexpr int SDIM = 2048;
static constexpr int EDIM = 1024;

static constexpr int BM = 128;
static constexpr int BN = 128;
static constexpr int BK = 32;
static constexpr int BK_PAD = 48;   // 96B row stride (16B aligned)
static constexpr int BN_PAD = 144;  // 288B row stride (16B aligned)

// ---------------------------------------------------------------------------
// Async 16B global -> LDS copy (no VGPR staging, tracked by ASYNCcnt)
// Toolchain's builtin takes int4* pointers (per clang diagnostic).
// ---------------------------------------------------------------------------
__device__ __forceinline__ void async_b128(const bf16* g, bf16* l) {
#if __has_builtin(__builtin_amdgcn_global_load_async_to_lds_b128)
    __builtin_amdgcn_global_load_async_to_lds_b128(
        (v4i_vs*)g, (v4i_vs*)l, 0, 0);
#else
    asm volatile("global_load_async_to_lds_b128 %0, %1, off"
                 :: "v"((unsigned)(unsigned long long)l), "v"(g)
                 : "memory");
#endif
}

__device__ __forceinline__ void wait_async_zero() {
#if __has_builtin(__builtin_amdgcn_s_wait_asynccnt)
    __builtin_amdgcn_s_wait_asynccnt(0);
#else
    asm volatile("s_wait_asynccnt 0x0" ::: "memory");
#endif
}

// ---------------------------------------------------------------------------
// fp32 -> bf16 conversion (grid-stride)
// ---------------------------------------------------------------------------
__global__ __launch_bounds__(256)
void cvt_f32_bf16(const float* __restrict__ in, bf16* __restrict__ out, long n) {
    long i = (long)blockIdx.x * 256 + threadIdx.x;
    long stride = (long)gridDim.x * 256;
    for (; i < n; i += stride) out[i] = (__bf16)in[i];
}

// ---------------------------------------------------------------------------
// Tiled bf16 WMMA GEMM.  C[m,n] = scale * sum_k A[m,k]*B(k,n) (+ bias[n])
//   B_TRANS=true : B stored row-major [N,K]   (weights W[n,k], or k^T case)
//   B_TRANS=false: B stored row-major [K,N]   (value case)
//   A_SCRAMBLE   : A[m,k] = Asrc[((m*E+k)%S)*E + ((m*E+k)/S)]  (view scramble)
// 256 threads = 8 waves, waves arranged 4(M) x 2(N), each wave owns a
// 32x64 region = 2x4 tiles of v_wmma_f32_16x16x32_bf16.
// Async double-buffered LDS: tile k+1's DMA overlaps tile k's WMMA.
// All M,N multiples of 128 and K multiples of 32 here -> no bounds checks.
// ---------------------------------------------------------------------------
template<bool A_SCRAMBLE, bool B_TRANS, bool HAS_BIAS, bool OUT_BF16>
__global__ __launch_bounds__(256)
void wmma_gemm(const bf16* __restrict__ A, const bf16* __restrict__ Bm,
               const float* __restrict__ bias, float scale,
               void* __restrict__ Cp,
               int M, int N, int K,
               long aBatch, long bBatch, long cBatch)
{
    const int batch = blockIdx.z;
    const int bm0 = blockIdx.y * BM;
    const int bn0 = blockIdx.x * BN;
    const bf16* Ab = A  + (long)batch * aBatch;
    const bf16* Bb = Bm + (long)batch * bBatch;

    __shared__ bf16 As[2][BM][BK_PAD];
    constexpr int BSR = B_TRANS ? BN : BK;
    constexpr int BSC = B_TRANS ? BK_PAD : BN_PAD;
    __shared__ bf16 Bs[2][BSR][BSC];

    const int tid  = threadIdx.x;
    const int lane = tid & 31;
    const int wave = tid >> 5;
    const int wm = wave & 3;    // 4 waves along M -> 32 rows each
    const int wn = wave >> 2;   // 2 waves along N -> 64 cols each

    // Issue async (or gather) copies for one K tile into LDS buffer `buf`.
    auto issue_tile = [&](int k0, int buf) {
        // ---- A tile: 128 rows x 32 k (512 slots of 8 bf16 = 16B) ----
#pragma unroll
        for (int rep = 0; rep < 2; ++rep) {
            int slot = tid + rep * 256;
            int row  = slot >> 2;
            int c8   = (slot & 3) * 8;
            if constexpr (A_SCRAMBLE) {
#pragma unroll
                for (int j = 0; j < 8; ++j) {
                    long flat = (long)(bm0 + row) * EDIM + (k0 + c8 + j);
                    int ss = (int)(flat & (SDIM - 1));
                    int ee = (int)(flat >> 11);       // / SDIM
                    As[buf][row][c8 + j] = Ab[(long)ss * EDIM + ee];
                }
            } else {
                async_b128(Ab + (long)(bm0 + row) * K + (k0 + c8),
                           &As[buf][row][c8]);
            }
        }
        // ---- B tile ----
        if constexpr (B_TRANS) {            // [128 n][32 k]
#pragma unroll
            for (int rep = 0; rep < 2; ++rep) {
                int slot = tid + rep * 256;
                int row  = slot >> 2;
                int c8   = (slot & 3) * 8;
                async_b128(Bb + (long)(bn0 + row) * K + (k0 + c8),
                           &Bs[buf][row][c8]);
            }
        } else {                             // [32 k][128 n]
#pragma unroll
            for (int rep = 0; rep < 2; ++rep) {
                int slot = tid + rep * 256;
                int row  = slot >> 4;
                int c8   = (slot & 15) * 8;
                async_b128(Bb + (long)(k0 + row) * N + (bn0 + c8),
                           &Bs[buf][row][c8]);
            }
        }
    };

    v8f acc[2][4];
#pragma unroll
    for (int i = 0; i < 2; ++i)
#pragma unroll
        for (int j = 0; j < 4; ++j)
            acc[i][j] = (v8f){0.f,0.f,0.f,0.f,0.f,0.f,0.f,0.f};

    issue_tile(0, 0);
    int cur = 0;
    for (int k0 = 0; k0 < K; k0 += BK) {
        // Tile k landed for this wave; barrier => landed for all waves AND
        // all waves finished reading the buffer we are about to refill.
        wait_async_zero();
        __syncthreads();
        if (k0 + BK < K) issue_tile(k0 + BK, cur ^ 1);   // overlap DMA w/ WMMA

        // ------------------ build fragments from buffer `cur` -----------
        // A frag (16x32 bf16): lanes 0-15 -> M rows, lane>>4 selects K half;
        // elems 0..15 -> K=base..base+15 (base = (lane>>4)*16).
        v16bf afr[2];
        {
            int am  = lane & 15;
            int akg = (lane >> 4) * 16;
#pragma unroll
            for (int t = 0; t < 2; ++t) {
                int arow = wm * 32 + t * 16 + am;
#pragma unroll
                for (int e = 0; e < 16; ++e) {
                    int kk = akg + e;
                    afr[t][e] = As[cur][arow][kk];
                }
            }
        }
        // B frag (32x16 bf16): N = lane&15, lane>>4 selects K half (16 rows).
        v16bf bfr[4];
        {
            int bnl = lane & 15;
            int bk0 = (lane >> 4) * 16;
#pragma unroll
            for (int t = 0; t < 4; ++t) {
                int bcol = wn * 64 + t * 16 + bnl;
#pragma unroll
                for (int e = 0; e < 16; ++e) {
                    if constexpr (B_TRANS) bfr[t][e] = Bs[cur][bcol][bk0 + e];
                    else                   bfr[t][e] = Bs[cur][bk0 + e][bcol];
                }
            }
        }
#pragma unroll
        for (int i = 0; i < 2; ++i)
#pragma unroll
            for (int j = 0; j < 4; ++j)
                acc[i][j] = __builtin_amdgcn_wmma_f32_16x16x32_bf16(
                    false, afr[i], false, bfr[j], (short)0, acc[i][j],
                    false, false);
        cur ^= 1;
    }

    // ------------------ store C ------------------
    // C layout: VGPR r: lanes 0-15 -> M=r, lanes 16-31 -> M=8+r; N = lane&15.
    {
        const long cbase = (long)batch * cBatch;
        const int cn_l = lane & 15;
        const int mr   = (lane >> 4) * 8;
#pragma unroll
        for (int i = 0; i < 2; ++i) {
#pragma unroll
            for (int j = 0; j < 4; ++j) {
                int n = bn0 + wn * 64 + j * 16 + cn_l;
                float bv = 0.f;
                if constexpr (HAS_BIAS) bv = bias[n];
#pragma unroll
                for (int r = 0; r < 8; ++r) {
                    int m = bm0 + wm * 32 + i * 16 + mr + r;
                    float v = acc[i][j][r] * scale + bv;
                    if constexpr (OUT_BF16)
                        ((bf16*)Cp)[cbase + (long)m * N + n] = (__bf16)v;
                    else
                        ((float*)Cp)[cbase + (long)m * N + n] = v;
                }
            }
        }
    }
}

// ---------------------------------------------------------------------------
// Row softmax over S=2048: one block per row, 8 elems/thread in registers.
// Writes f32 attn (model output) and bf16 attn (for the A*V WMMA GEMM).
// ---------------------------------------------------------------------------
__global__ __launch_bounds__(256)
void softmax_rows(float* __restrict__ attn, bf16* __restrict__ attn_b)
{
    const long row = blockIdx.x;               // B*S rows
    float* p = attn + row * (long)SDIM;
    bf16*  q = attn_b + row * (long)SDIM;
    const int t = threadIdx.x;

    float v[8];
    float mx = -3.402823466e+38f;
#pragma unroll
    for (int j = 0; j < 8; ++j) {
        v[j] = p[t + j * 256];
        mx = fmaxf(mx, v[j]);
    }
#pragma unroll
    for (int off = 16; off > 0; off >>= 1)
        mx = fmaxf(mx, __shfl_xor(mx, off, 32));

    __shared__ float smax[8];
    __shared__ float ssum[8];
    if ((t & 31) == 0) smax[t >> 5] = mx;
    __syncthreads();
    float m2 = smax[0];
#pragma unroll
    for (int i = 1; i < 8; ++i) m2 = fmaxf(m2, smax[i]);

    float s = 0.f;
#pragma unroll
    for (int j = 0; j < 8; ++j) {
        v[j] = __expf(v[j] - m2);
        s += v[j];
    }
#pragma unroll
    for (int off = 16; off > 0; off >>= 1)
        s += __shfl_xor(s, off, 32);
    if ((t & 31) == 0) ssum[t >> 5] = s;
    __syncthreads();
    float tot = 0.f;
#pragma unroll
    for (int i = 0; i < 8; ++i) tot += ssum[i];
    float inv = 1.0f / tot;

#pragma unroll
    for (int j = 0; j < 8; ++j) {
        float a = v[j] * inv;
        p[t + j * 256] = a;
        q[t + j * 256] = (__bf16)a;
    }
}

// ---------------------------------------------------------------------------
// Host-side orchestration
// ---------------------------------------------------------------------------
extern "C" void kernel_launch(void* const* d_in, const int* in_sizes, int n_in,
                              void* d_out, int out_size, void* d_ws, size_t ws_size,
                              hipStream_t stream) {
    (void)in_sizes; (void)n_in; (void)out_size; (void)ws_size;

    const float* query = (const float*)d_in[0];
    const float* key   = (const float*)d_in[1];
    const float* value = (const float*)d_in[2];
    const float* Wq    = (const float*)d_in[3];
    const float* bq    = (const float*)d_in[4];
    const float* Wk    = (const float*)d_in[5];
    const float* bk    = (const float*)d_in[6];
    const float* Wo    = (const float*)d_in[7];
    const float* bo    = (const float*)d_in[8];

    const long nBSE = (long)BDIM * SDIM * EDIM;   // 8,388,608
    const long nEE  = (long)EDIM * EDIM;          // 1,048,576
    const long nSE  = (long)SDIM * EDIM;

    // workspace layout (bf16); attn_b aliases xq+xk (both exactly 32MB, and
    // xq/xk are dead before softmax runs)
    bf16* xq  = (bf16*)d_ws;
    bf16* xk  = xq  + nBSE;
    bf16* vb  = xk  + nBSE;
    bf16* qb  = vb  + nBSE;
    bf16* kb  = qb  + nBSE;
    bf16* avb = kb  + nBSE;
    bf16* wqb = avb + nBSE;
    bf16* wkb = wqb + nEE;
    bf16* wob = wkb + nEE;
    bf16* attn_b = xq;          // reuse: B*S*S == 2 * nBSE elements

    float* outp  = (float*)d_out;          // [B,S,E]
    float* attnp = outp + nBSE;            // [B,S,S]

    // 1) fp32 -> bf16 conversions
    cvt_f32_bf16<<<2048, 256, 0, stream>>>(query, xq, nBSE);
    cvt_f32_bf16<<<2048, 256, 0, stream>>>(key,   xk, nBSE);
    cvt_f32_bf16<<<2048, 256, 0, stream>>>(value, vb, nBSE);
    cvt_f32_bf16<<<512,  256, 0, stream>>>(Wq, wqb, nEE);
    cvt_f32_bf16<<<512,  256, 0, stream>>>(Wk, wkb, nEE);
    cvt_f32_bf16<<<512,  256, 0, stream>>>(Wo, wob, nEE);

    // 2) projections: q = query@Wq^T + bq ; k = key@Wk^T + bk  (M=8192)
    {
        dim3 grid(EDIM / BN, (BDIM * SDIM) / BM, 1);
        wmma_gemm<false, true, true, true><<<grid, 256, 0, stream>>>(
            xq, wqb, bq, 1.0f, qb, BDIM * SDIM, EDIM, EDIM, 0, 0, 0);
        wmma_gemm<false, true, true, true><<<grid, 256, 0, stream>>>(
            xk, wkb, bk, 1.0f, kb, BDIM * SDIM, EDIM, EDIM, 0, 0, 0);
    }

    // 3) scores = q @ k^T / sqrt(E)   -> f32 attn region of d_out
    {
        dim3 grid(SDIM / BN, SDIM / BM, BDIM);
        wmma_gemm<false, true, false, false><<<grid, 256, 0, stream>>>(
            qb, kb, nullptr, 1.0f / 32.0f, attnp,
            SDIM, SDIM, EDIM, nSE, nSE, (long)SDIM * SDIM);
    }

    // 4) softmax rows (writes f32 output + bf16 copy for the AV GEMM)
    softmax_rows<<<BDIM * SDIM, 256, 0, stream>>>(attnp, attn_b);

    // 5) av = attn @ value   (B stored [K=S, N=E])
    {
        dim3 grid(EDIM / BN, SDIM / BM, BDIM);
        wmma_gemm<false, false, false, true><<<grid, 256, 0, stream>>>(
            attn_b, vb, nullptr, 1.0f, avb,
            SDIM, EDIM, SDIM, (long)SDIM * SDIM, nSE, nSE);
    }

    // 6) out = scramble(av) @ Wo^T + bo  -> f32 d_out
    {
        dim3 grid(EDIM / BN, SDIM / BM, BDIM);
        wmma_gemm<true, true, true, false><<<grid, 256, 0, stream>>>(
            avb, wob, bo, 1.0f, outp,
            SDIM, EDIM, EDIM, nSE, 0, nSE);
    }
}
